// twoway_concat_52158082843418
// MI455X (gfx1250) — matrix operand order repaired and checked by reading, compiled, not run on hardware
//
#include <hip/hip_runtime.h>
#include <math.h>

// ---------------- problem constants ----------------
constexpr int B_  = 4096;
constexpr int K_  = 100;
constexpr int NU_ = 100000;
constexpr int NI_ = 50000;
constexpr int M_  = B_ * K_;      // 409600 rows through the MLPs

// WMMA vector types (CDNA5 gfx1250, wave32)
typedef __attribute__((ext_vector_type(8)))  __bf16 v8bf;
typedef __attribute__((ext_vector_type(16))) __bf16 v16bf;
typedef __attribute__((ext_vector_type(8)))  float  v8f;

// ---------------- kernel A: 64-wide row sums of embedding tables ----------------
__global__ void rowsum64_kernel(const float* __restrict__ emb, float* __restrict__ out, int n) {
  int i = blockIdx.x * 256 + threadIdx.x;
  if (i >= n) return;
  const float4* p = (const float4*)(emb + (size_t)i * 64);
  float s = 0.f;
#pragma unroll
  for (int j = 0; j < 16; ++j) {
    float4 v = p[j];
    s += (v.x + v.y) + (v.z + v.w);
  }
  out[i] = s;
}

// ---------------- kernel B: gather neighbor row-sums + scores per (b,k) ----------------
__global__ void gather_kernel(const int* __restrict__ uix, const int* __restrict__ iix,
                              const int* __restrict__ unbr, const int* __restrict__ inbr,
                              const float* __restrict__ uscr, const float* __restrict__ iscr,
                              const float* __restrict__ Su, const float* __restrict__ Si,
                              float* __restrict__ r_u, float* __restrict__ u_sg,
                              float* __restrict__ r_i, float* __restrict__ i_sg) {
  int idx = blockIdx.x * 256 + threadIdx.x;
  if (idx >= M_) return;
  int b = idx / K_, k = idx - b * K_;
  int u = uix[b], it = iix[b];
  r_u[idx]  = Su[unbr[(size_t)u * K_ + k]];
  r_i[idx]  = Si[inbr[(size_t)it * K_ + k]];
  u_sg[idx] = uscr[(size_t)u * K_ + k];
  i_sg[idx] = iscr[(size_t)it * K_ + k];
}

// ---------------- kernel C: rank-1 layer-1 projections: [B,100]@[100,100] x4 ----------------
// p1u = r_u @ n1W1_top, p1i = r_i @ n1W1_bot, p2u = u_s @ n2W1_top, p2i = i_s @ n2W1_bot
__global__ __launch_bounds__(128) void proj_kernel(
    const float* __restrict__ r_u, const float* __restrict__ u_sg,
    const float* __restrict__ r_i, const float* __restrict__ i_sg,
    const float* __restrict__ n1w1, const float* __restrict__ n2w1,
    float* __restrict__ p1u, float* __restrict__ p1i,
    float* __restrict__ p2u, float* __restrict__ p2i) {
  __shared__ float sru[K_], sus[K_], sri[K_], sis[K_];
  const int b = blockIdx.x;
  const int j = threadIdx.x;
  if (j < K_) {
    sru[j] = r_u[(size_t)b * K_ + j];
    sus[j] = u_sg[(size_t)b * K_ + j];
    sri[j] = r_i[(size_t)b * K_ + j];
    sis[j] = i_sg[(size_t)b * K_ + j];
  }
  __syncthreads();
  if (j >= K_) return;
  float a1 = 0.f, a2 = 0.f, a3 = 0.f, a4 = 0.f;
  for (int k = 0; k < K_; ++k) {
    float w1t = n1w1[(size_t)k * K_ + j];          // n1_w1 is [200,100] row-major
    float w1b = n1w1[(size_t)(K_ + k) * K_ + j];
    float w2t = n2w1[(size_t)k * K_ + j];
    float w2b = n2w1[(size_t)(K_ + k) * K_ + j];
    a1 = fmaf(sru[k], w1t, a1);
    a2 = fmaf(sri[k], w1b, a2);
    a3 = fmaf(sus[k], w2t, a3);
    a4 = fmaf(sis[k], w2b, a4);
  }
  p1u[(size_t)b * K_ + j] = a1;
  p1i[(size_t)b * K_ + j] = a2;
  p2u[(size_t)b * K_ + j] = a3;
  p2i[(size_t)b * K_ + j] = a4;
}

// ---------------- weight pack: f32 [K,N] -> bf16 per-lane WMMA B-operand layout ----------------
// flat = ((ntile*(Kp/32) + kchunk)*32 + lane)*16 + e
// lane<16: column n = ntile*16+lane, K = kchunk*32 + (e<8 ? e : 16+e-8)
// lane>=16: same column group, K = kchunk*32 + 8 + (e<8 ? e : 16+e-8)
__global__ void packB_kernel(const float* __restrict__ W, int Kd, int Nd, int Kp, int Np,
                             __bf16* __restrict__ out) {
  int total = (Np / 16) * (Kp / 32) * 512;
  int idx = blockIdx.x * 256 + threadIdx.x;
  if (idx >= total) return;
  int e  = idx & 15;
  int l  = (idx >> 4) & 31;
  int cc = idx >> 9;                 // ntile*(Kp/32) + kchunk
  int kchunks = Kp >> 5;
  int kch = cc % kchunks;
  int t   = cc / kchunks;
  int n = t * 16 + (l & 15);
  int k = kch * 32 + ((l >> 4) * 8) + (e & 7) + ((e >> 3) * 16);
  float v = (k < Kd && n < Nd) ? W[(size_t)k * Nd + n] : 0.f;
  out[idx] = (__bf16)v;
}

// ---------------- zero accumulator ----------------
__global__ void zero_kernel(float* __restrict__ p, int n) {
  int i = blockIdx.x * 256 + threadIdx.x;
  if (i < n) p[i] = 0.f;
}

// ---------------- WMMA helpers ----------------
// A-operand (16-bit, 16x32): lane l holds row m=l&15; VGPR0-3 = K[half*8 .. +7], VGPR4-7 = K[half*8+16 .. +23]
__device__ __forceinline__ v16bf ldsA(const __bf16* h, int stride, int m, int half, int k0) {
  const __bf16* p = h + m * stride + k0 + half * 8;
  v8bf lo = *(const v8bf*)p;
  v8bf hi = *(const v8bf*)(p + 16);
  return __builtin_shufflevector(lo, hi, 0, 1, 2, 3, 4, 5, 6, 7, 8, 9, 10, 11, 12, 13, 14, 15);
}

// B-operand prepacked: 16 contiguous bf16 per lane per (ntile,kchunk)
__device__ __forceinline__ v16bf ldB(const __bf16* wpk, int lane) {
  const __bf16* p = wpk + lane * 16;
  v8bf lo = *(const v8bf*)p;
  v8bf hi = *(const v8bf*)(p + 8);
  return __builtin_shufflevector(lo, hi, 0, 1, 2, 3, 4, 5, 6, 7, 8, 9, 10, 11, 12, 13, 14, 15);
}

// One MLP layer on a 16-row tile: hin [16,kp] bf16 (LDS) @ packed W [kp,np] -> relu -> hout [16,np]
__device__ __forceinline__ void wmma_layer(const __bf16* hin, int kp, const __bf16* wpk,
                                           const float* bias, int nvalid, int np,
                                           __bf16* hout, int lane) {
  const int m = lane & 15;
  const int half = lane >> 4;
  const int kch = kp >> 5;
  const int ntl = np >> 4;
  const int mb = half * 8;           // C/D layout: VGPR i -> row i + 8*half
  for (int t = 0; t < ntl; ++t) {
    v8f c = {0.f, 0.f, 0.f, 0.f, 0.f, 0.f, 0.f, 0.f};
    for (int cc = 0; cc < kch; ++cc) {
      v16bf a = ldsA(hin, kp, m, half, cc * 32);
      v16bf b = ldB(wpk + (size_t)(t * kch + cc) * 512, lane);
      c = __builtin_amdgcn_wmma_f32_16x16x32_bf16(false, a, false, b, (short)0, c, false, false);
    }
    int n = t * 16 + m;              // C/D layout: N = lane&15
    float bv = (n < nvalid) ? bias[n] : 0.f;
#pragma unroll
    for (int i = 0; i < 8; ++i) {
      float v = c[i] + bv;
      v = v > 0.f ? v : 0.f;
      hout[(mb + i) * np + n] = (__bf16)v;
    }
  }
}

// ---------------- fused MLP: one wave per 16 rows, all layers in LDS/regs ----------------
__global__ __launch_bounds__(256) void mlp_fused_kernel(
    const float* __restrict__ r_u, const float* __restrict__ u_sg,
    const float* __restrict__ r_i, const float* __restrict__ i_sg,
    const float* __restrict__ p1u, const float* __restrict__ p1i,
    const float* __restrict__ p2u, const float* __restrict__ p2i,
    const __bf16* __restrict__ w2p1, const __bf16* __restrict__ w3p1, const __bf16* __restrict__ w4p1,
    const __bf16* __restrict__ w2p2, const __bf16* __restrict__ w3p2, const __bf16* __restrict__ w4p2,
    const float* __restrict__ n1b1, const float* __restrict__ n1b2, const float* __restrict__ n1b3,
    const float* __restrict__ n1b4, const float* __restrict__ n1w5, const float* __restrict__ n1b5,
    const float* __restrict__ n2b1, const float* __restrict__ n2b2, const float* __restrict__ n2b3,
    const float* __restrict__ n2b4, const float* __restrict__ n2w5, const float* __restrict__ n2b5,
    float* __restrict__ acc) {
  __shared__ __align__(16) __bf16 shA[8][16 * 128];   // 32 KB
  __shared__ __align__(16) __bf16 shB[8][16 * 64];    // 16 KB
  const int wave = threadIdx.x >> 5;
  const int lane = threadIdx.x & 31;
  const int tile = (blockIdx.x * 8 + wave) * 16;      // global row base (M_ % 128 == 0)
  __bf16* HA = shA[wave];
  __bf16* HB = shB[wave];
  float csum = 0.f;                                   // valid on lanes 0..15 (per-row c1+c2)

  for (int net = 0; net < 2; ++net) {
    // net0 (MLP1 on x1): coefs (u_s, i_s), projections (p1u, p1i)
    // net1 (MLP2 on x2): coefs (r_u, r_i), projections (p2u, p2i)
    const float* cu = net ? r_u : u_sg;
    const float* ci = net ? r_i : i_sg;
    const float* pu = net ? p2u : p1u;
    const float* pi = net ? p2i : p1i;
    const __bf16* W2 = net ? w2p2 : w2p1;
    const __bf16* W3 = net ? w3p2 : w3p1;
    const __bf16* W4 = net ? w4p2 : w4p1;
    const float* Bi1 = net ? n2b1 : n1b1;
    const float* Bi2 = net ? n2b2 : n1b2;
    const float* Bi3 = net ? n2b3 : n1b3;
    const float* Bi4 = net ? n2b4 : n1b4;
    const float* W5  = net ? n2w5 : n1w5;
    const float* Bi5 = net ? n2b5 : n1b5;

    // ---- layer 1 (rank-1 reconstruction): h1[16][128] bf16, zero-padded cols 100..127
    for (int r = 0; r < 16; ++r) {
      const int grow = tile + r;
      const int b = grow / K_;
      const float a0 = cu[grow];
      const float a1 = ci[grow];
      const float* purow = pu + (size_t)b * K_;
      const float* pirow = pi + (size_t)b * K_;
      for (int j = lane; j < 128; j += 32) {
        float v = 0.f;
        if (j < K_) {
          v = fmaf(a0, purow[j], fmaf(a1, pirow[j], Bi1[j]));
          v = v > 0.f ? v : 0.f;
        }
        HA[r * 128 + j] = (__bf16)v;
      }
    }
    __syncthreads();
    // ---- layer 2: [16,128]@[128,64] (valid 100x50)
    wmma_layer(HA, 128, W2, Bi2, 50, 64, HB, lane);
    __syncthreads();
    // ---- layer 3: [16,64]@[64,32] (valid 50x25)
    wmma_layer(HB, 64, W3, Bi3, 25, 32, HA, lane);
    __syncthreads();
    // ---- layer 4: [16,32]@[32,16] (valid 25x12)
    wmma_layer(HA, 32, W4, Bi4, 12, 16, HB, lane);
    __syncthreads();
    // ---- layer 5: per-row dot(12) + bias (no relu)
    if (lane < 16) {
      const __bf16* hr = HB + lane * 16;
      float s = Bi5[0];
#pragma unroll
      for (int j = 0; j < 12; ++j) s = fmaf((float)hr[j], W5[j], s);
      csum += s;
    }
    __syncthreads();
  }

  if (lane < 16) {
    const int grow = tile + lane;
    atomicAdd(&acc[grow / K_], csum);   // global_atomic_add_f32
  }
}

// ---------------- finalize: sigmoid(mean over K) ----------------
__global__ void finalize_kernel(const float* __restrict__ acc, float* __restrict__ out) {
  int b = blockIdx.x * 256 + threadIdx.x;
  if (b < B_) out[b] = 1.f / (1.f + expf(-acc[b] * (1.f / (float)K_)));
}

// ---------------- host launcher ----------------
extern "C" void kernel_launch(void* const* d_in, const int* in_sizes, int n_in,
                              void* d_out, int out_size, void* d_ws, size_t ws_size,
                              hipStream_t stream) {
  (void)in_sizes; (void)n_in; (void)out_size; (void)ws_size;
  const int*   user_idxs = (const int*)d_in[0];
  const int*   item_idxs = (const int*)d_in[1];
  const int*   u_nbr     = (const int*)d_in[2];
  const int*   i_nbr     = (const int*)d_in[3];
  const float* u_scr     = (const float*)d_in[4];
  const float* i_scr     = (const float*)d_in[5];
  const float* u_emb     = (const float*)d_in[6];
  const float* i_emb     = (const float*)d_in[7];
  const float* n1w1 = (const float*)d_in[8];
  const float* n1b1 = (const float*)d_in[9];
  const float* n1w2 = (const float*)d_in[10];
  const float* n1b2 = (const float*)d_in[11];
  const float* n1w3 = (const float*)d_in[12];
  const float* n1b3 = (const float*)d_in[13];
  const float* n1w4 = (const float*)d_in[14];
  const float* n1b4 = (const float*)d_in[15];
  const float* n1w5 = (const float*)d_in[16];
  const float* n1b5 = (const float*)d_in[17];
  const float* n2w1 = (const float*)d_in[18];
  const float* n2b1 = (const float*)d_in[19];
  const float* n2w2 = (const float*)d_in[20];
  const float* n2b2 = (const float*)d_in[21];
  const float* n2w3 = (const float*)d_in[22];
  const float* n2b3 = (const float*)d_in[23];
  const float* n2w4 = (const float*)d_in[24];
  const float* n2b4 = (const float*)d_in[25];
  const float* n2w5 = (const float*)d_in[26];
  const float* n2b5 = (const float*)d_in[27];
  float* out = (float*)d_out;

  // workspace carve-up (256B-aligned)
  char* ws = (char*)d_ws;
  size_t off = 0;
  auto alloc = [&](size_t bytes) -> char* {
    char* p = ws + off;
    off = (off + bytes + 255) & ~(size_t)255;
    return p;
  };
  float* Su   = (float*)alloc((size_t)NU_ * 4);
  float* Si   = (float*)alloc((size_t)NI_ * 4);
  float* r_u  = (float*)alloc((size_t)M_ * 4);
  float* u_sg = (float*)alloc((size_t)M_ * 4);
  float* r_i  = (float*)alloc((size_t)M_ * 4);
  float* i_sg = (float*)alloc((size_t)M_ * 4);
  float* p1u  = (float*)alloc((size_t)M_ * 4);
  float* p1i  = (float*)alloc((size_t)M_ * 4);
  float* p2u  = (float*)alloc((size_t)M_ * 4);
  float* p2i  = (float*)alloc((size_t)M_ * 4);
  float* acc  = (float*)alloc((size_t)B_ * 4);
  constexpr int PK2 = (64 / 16) * (128 / 32) * 512;  // 8192 bf16
  constexpr int PK3 = (32 / 16) * (64 / 32) * 512;   // 2048
  constexpr int PK4 = (16 / 16) * (32 / 32) * 512;   // 512
  __bf16* w2p1 = (__bf16*)alloc(PK2 * 2);
  __bf16* w3p1 = (__bf16*)alloc(PK3 * 2);
  __bf16* w4p1 = (__bf16*)alloc(PK4 * 2);
  __bf16* w2p2 = (__bf16*)alloc(PK2 * 2);
  __bf16* w3p2 = (__bf16*)alloc(PK3 * 2);
  __bf16* w4p2 = (__bf16*)alloc(PK4 * 2);

  // A: embedding row-sums (the only touch of the big embedding tables)
  rowsum64_kernel<<<(NU_ + 255) / 256, 256, 0, stream>>>(u_emb, Su, NU_);
  rowsum64_kernel<<<(NI_ + 255) / 256, 256, 0, stream>>>(i_emb, Si, NI_);
  // B: gathers
  gather_kernel<<<(M_ + 255) / 256, 256, 0, stream>>>(user_idxs, item_idxs, u_nbr, i_nbr,
                                                      u_scr, i_scr, Su, Si,
                                                      r_u, u_sg, r_i, i_sg);
  // C: rank-1 layer-1 projections
  proj_kernel<<<B_, 128, 0, stream>>>(r_u, u_sg, r_i, i_sg, n1w1, n2w1, p1u, p1i, p2u, p2i);
  // weight packs (tiny)
  packB_kernel<<<(PK2 + 255) / 256, 256, 0, stream>>>(n1w2, 100, 50, 128, 64, w2p1);
  packB_kernel<<<(PK3 + 255) / 256, 256, 0, stream>>>(n1w3, 50, 25, 64, 32, w3p1);
  packB_kernel<<<(PK4 + 255) / 256, 256, 0, stream>>>(n1w4, 25, 12, 32, 16, w4p1);
  packB_kernel<<<(PK2 + 255) / 256, 256, 0, stream>>>(n2w2, 100, 50, 128, 64, w2p2);
  packB_kernel<<<(PK3 + 255) / 256, 256, 0, stream>>>(n2w3, 50, 25, 64, 32, w3p2);
  packB_kernel<<<(PK4 + 255) / 256, 256, 0, stream>>>(n2w4, 25, 12, 32, 16, w4p2);
  // accumulator init (fresh every launch: deterministic)
  zero_kernel<<<(B_ + 255) / 256, 256, 0, stream>>>(acc, B_);
  // D: fused WMMA MLP over 409600 rows, 8 waves/block, 16 rows/wave
  mlp_fused_kernel<<<M_ / 128, 256, 0, stream>>>(
      r_u, u_sg, r_i, i_sg, p1u, p1i, p2u, p2i,
      w2p1, w3p1, w4p1, w2p2, w3p2, w4p2,
      n1b1, n1b2, n1b3, n1b4, n1w5, n1b5,
      n2b1, n2b2, n2b3, n2b4, n2w5, n2b5, acc);
  // E: sigmoid(mean)
  finalize_kernel<<<(B_ + 255) / 256, 256, 0, stream>>>(acc, out);
}
